// Encoding_1039382085847
// MI455X (gfx1250) — matrix operand order repaired and checked by reading, not compiled, not run on hardware
//
#include <hip/hip_runtime.h>

#define B_ 32
#define N_ 4096
#define D_ 256
#define K_ 128
#define NT 64   // n-rows per tile

typedef __attribute__((ext_vector_type(16))) _Float16 v16h;
typedef __attribute__((ext_vector_type(8)))  float  v8f;
typedef __attribute__((ext_vector_type(4)))  unsigned int u32x4;
typedef __attribute__((ext_vector_type(2)))  unsigned int u32x2;
typedef __attribute__((ext_vector_type(4)))  float  f32x4;

union FragU { u32x4 q[2]; v16h v; };
union Pack4 { _Float16 h[4]; u32x2 u; };

// ---------------- zero wsum ----------------
__global__ void zero_wsum(float* __restrict__ wsum) {
  int i = blockIdx.x * 256 + threadIdx.x;
  if (i < B_ * K_) wsum[i] = 0.f;
}

// ---------------- prep: codewords f32 -> f16, c2[k] = ||c_k||^2 -------------
__global__ __launch_bounds__(256) void prep_cw(const float* __restrict__ cw,
                                               _Float16* __restrict__ cwh,
                                               float* __restrict__ c2) {
  __shared__ float red[256];
  int k = blockIdx.x, d = threadIdx.x;
  float v = cw[k * D_ + d];
  cwh[k * D_ + d] = (_Float16)v;
  red[d] = v * v;
  __syncthreads();
  for (int s = 128; s > 0; s >>= 1) {
    if (d < s) red[d] += red[d + s];
    __syncthreads();
  }
  if (d == 0) c2[k] = red[0];
}

// ---- fused main: loop n-tiles, GEMM1 -> softmax -> GEMM2 (reg-accumulated) -
__global__ __launch_bounds__(256) void encode_main(
    const float* __restrict__ x, const float* __restrict__ scale,
    const _Float16* __restrict__ cwh, const float* __restrict__ c2g,
    float* __restrict__ partial, float* __restrict__ wsum, int C, int T) {

  __shared__ __align__(16) _Float16 xs [NT * D_];   // 32 KB x f16 row-major
  __shared__ __align__(16) _Float16 xts[D_ * NT];   // 32 KB x f16 transposed
  __shared__ __align__(16) float    xcs[NT * K_];   // 32 KB x.c^T f32
  __shared__ __align__(16) _Float16 wts[K_ * NT];   // 16 KB W transposed f16
  __shared__ float x2s[NT];
  __shared__ float red[256];
  __shared__ float scl[K_], c2l[K_];

  const int t    = threadIdx.x;
  const int wave = t >> 5;
  const int lane = t & 31;
  const int b     = blockIdx.x / C;
  const int chunk = blockIdx.x - b * C;

  if (t < K_) { scl[t] = scale[t]; c2l[t] = c2g[t]; }

  // persistent GEMM2 accumulators: wave owns k-rows [16*wave, 16*wave+16),
  // 16 d-tiles of 16x16 -> 128 f32 per lane
  v8f acc2[16];
#pragma unroll
  for (int i = 0; i < 16; ++i) acc2[i] = {};
  float wsum_loc = 0.f;

  const float* xbase = x + ((size_t)b * N_ + (size_t)chunk * T * NT) * D_;

  for (int it = 0; it < T; ++it) {
    const float* xp = xbase + (size_t)it * NT * D_;

    // ---- load x tile (non-temporal, streamed once): 4 threads/row ----
    {
      const int r = t & 63, q = t >> 6;
      const f32x4* src = (const f32x4*)(xp + r * D_ + q * 64);
      float ssq = 0.f;
#pragma unroll
      for (int i = 0; i < 16; ++i) {
        f32x4 v = __builtin_nontemporal_load(src + i);
        ssq += v.x * v.x + v.y * v.y + v.z * v.z + v.w * v.w;
        const int c = q * 64 + i * 4;
        Pack4 pk;
        pk.h[0] = (_Float16)v.x; pk.h[1] = (_Float16)v.y;
        pk.h[2] = (_Float16)v.z; pk.h[3] = (_Float16)v.w;
        *(u32x2*)&xs[r * D_ + c] = pk.u;
        xts[(c + 0) * NT + r] = pk.h[0];  xts[(c + 1) * NT + r] = pk.h[1];
        xts[(c + 2) * NT + r] = pk.h[2];  xts[(c + 3) * NT + r] = pk.h[3];
      }
      red[t] = ssq;
    }
    // prefetch next tile while we compute on this one
    if (it + 1 < T) {
      const char* np = (const char*)(xp + NT * D_) + t * 256;
      __builtin_prefetch(np, 0, 1);
      __builtin_prefetch(np + 128, 0, 1);
    }
    __syncthreads();
    if (t < NT) x2s[t] = red[t] + red[t + 64] + red[t + 128] + red[t + 192];
    __syncthreads();

    // ---- GEMM1: xc[64x128] = x_tile * cw^T (f16 WMMA, f32 acc) ----
    {
      const int mt  = wave & 3;
      const int kb4 = (wave >> 2) * 4;
#pragma unroll 2
      for (int kt = 0; kt < 4; ++kt) {
        const int ktile = kb4 + kt;
        v8f acc = {};
#pragma unroll
        for (int s = 0; s < 8; ++s) {
          FragU A, Bf;
          const int m  = mt * 16 + (lane & 15);
          const int kb = s * 32 + ((lane >> 4) << 3);
          A.q[0] = *(const u32x4*)&xs[m * D_ + kb];
          A.q[1] = *(const u32x4*)&xs[m * D_ + kb + 16];
          const int cr = ktile * 16 + (lane & 15);
          const int db = s * 32 + ((lane >> 4) << 3);
          const u32x4* cp = (const u32x4*)(cwh + cr * D_ + db);
          Bf.q[0] = cp[0];
          Bf.q[1] = cp[2];
          acc = __builtin_amdgcn_wmma_f32_16x16x32_f16(false, A.v, false, Bf.v,
                                                       (short)0, acc, false, false);
        }
        const int col  = ktile * 16 + (lane & 15);
        const int mrow = mt * 16 + ((lane >> 4) << 3);
#pragma unroll
        for (int v = 0; v < 8; ++v) xcs[(mrow + v) * K_ + col] = acc[v];
      }
    }
    __syncthreads();

    // ---- softmax over K per row; write W transposed f16 ----
    {
      const int r = t & 63, q = t >> 6;
      float sl[32];
      const float x2 = x2s[r];
      float m = -3.4e38f;
#pragma unroll
      for (int i = 0; i < 32; ++i) {
        const int k = q * 32 + i;
        const float v = scl[k] * (x2 - 2.f * xcs[r * K_ + k] + c2l[k]);
        sl[i] = v;
        m = fmaxf(m, v);
      }
      red[t] = m;
      __syncthreads();
      m = fmaxf(fmaxf(red[r], red[r + 64]), fmaxf(red[r + 128], red[r + 192]));
      __syncthreads();
      float s = 0.f;
#pragma unroll
      for (int i = 0; i < 32; ++i) { sl[i] = __expf(sl[i] - m); s += sl[i]; }
      red[t] = s;
      __syncthreads();
      s = red[r] + red[r + 64] + red[r + 128] + red[r + 192];
      const float inv = 1.0f / s;
#pragma unroll
      for (int i = 0; i < 32; ++i) {
        const int k = q * 32 + i;
        wts[k * NT + r] = (_Float16)(sl[i] * inv);
      }
    }
    __syncthreads();

    // ---- accumulate per-k sum of W locally ----
    if (t < K_) {
      float s = 0.f;
#pragma unroll 8
      for (int n = 0; n < NT; ++n) s += (float)wts[t * NT + n];
      wsum_loc += s;
    }

    // ---- GEMM2: acc2 += W^T (16x64 slice) * x_tile (64x256) ----
    {
      const int ktile = wave;
      FragU A0, A1;
      const int kr = ktile * 16 + (lane & 15);
      const int nb = (lane >> 4) << 3;
      A0.q[0] = *(const u32x4*)&wts[kr * NT + nb];
      A0.q[1] = *(const u32x4*)&wts[kr * NT + nb + 16];
      A1.q[0] = *(const u32x4*)&wts[kr * NT + 32 + nb];
      A1.q[1] = *(const u32x4*)&wts[kr * NT + 32 + nb + 16];
#pragma unroll
      for (int dt = 0; dt < 16; ++dt) {
        const int dr  = dt * 16 + (lane & 15);
        const int nb0 = (lane >> 4) << 3;
        FragU B0, B1;
        B0.q[0] = *(const u32x4*)&xts[dr * NT + nb0];
        B0.q[1] = *(const u32x4*)&xts[dr * NT + nb0 + 16];
        B1.q[0] = *(const u32x4*)&xts[dr * NT + 32 + nb0];
        B1.q[1] = *(const u32x4*)&xts[dr * NT + 32 + nb0 + 16];
        acc2[dt] = __builtin_amdgcn_wmma_f32_16x16x32_f16(false, A0.v, false, B0.v,
                                                          (short)0, acc2[dt], false, false);
        acc2[dt] = __builtin_amdgcn_wmma_f32_16x16x32_f16(false, A1.v, false, B1.v,
                                                          (short)0, acc2[dt], false, false);
      }
    }
    __syncthreads();   // protect xs/xts/wts before next tile's loads
  }

  // ---- one tiny atomic per (wg, k) for wsum ----
  if (t < K_) unsafeAtomicAdd(&wsum[b * K_ + t], wsum_loc);

  // ---- write partial E (exclusive ownership, plain stores) ----
  {
    float* pp = partial + ((size_t)blockIdx.x << 15);   // (b*C + chunk) * K*D
    const int ktile = wave;
    const int krow0 = ktile * 16 + ((lane >> 4) << 3);
#pragma unroll
    for (int dt = 0; dt < 16; ++dt) {
      float* ep = pp + (size_t)krow0 * D_ + dt * 16 + (lane & 15);
#pragma unroll
      for (int v = 0; v < 8; ++v) ep[(size_t)v * D_] = acc2[dt][v];
    }
  }
}

// ---- reduce partials + finalize: E = sum_c partial - wsum[b,k]*c[k,d] ------
__global__ void reduce_kernel(float* __restrict__ E, const float* __restrict__ partial,
                              const float* __restrict__ wsum, const float* __restrict__ cw,
                              int C, int n_out) {
  int i = blockIdx.x * 256 + threadIdx.x;
  if (i >= n_out) return;
  const int b = i >> 15, kd = i & (K_ * D_ - 1);
  const float* p = partial + (((size_t)b * C) << 15) + kd;
  float s = 0.f;
  for (int c = 0; c < C; ++c) s += p[(size_t)c << 15];
  E[i] = s - wsum[i >> 8] * cw[kd];
}

extern "C" void kernel_launch(void* const* d_in, const int* in_sizes, int n_in,
                              void* d_out, int out_size, void* d_ws, size_t ws_size,
                              hipStream_t stream) {
  const float* x     = (const float*)d_in[0];
  const float* cw    = (const float*)d_in[1];
  const float* scale = (const float*)d_in[2];
  float* E = (float*)d_out;

  // workspace layout: [cw f16: 64KB][c2: 512B][wsum: 16KB][partials: C*4MB]
  _Float16* cwh = (_Float16*)d_ws;
  const size_t c2_off   = (size_t)K_ * D_ * 2;
  const size_t wsum_off = c2_off + (size_t)K_ * 4;
  size_t part_off       = wsum_off + (size_t)B_ * K_ * 4;
  part_off = (part_off + 255) & ~(size_t)255;
  float* c2   = (float*)((char*)d_ws + c2_off);
  float* wsum = (float*)((char*)d_ws + wsum_off);

  const size_t batch_bytes = (size_t)B_ * K_ * D_ * 4;  // 4 MB of partials per chunk
  int C = 1;
  float* partial = E;   // C==1: each wg owns a whole batch, write d_out directly
  for (int c = 8; c >= 2; c >>= 1) {
    if (ws_size >= part_off + (size_t)c * batch_bytes) {
      C = c;
      partial = (float*)((char*)d_ws + part_off);
      break;
    }
  }
  const int T = (N_ / NT) / C;

  zero_wsum<<<(B_ * K_ + 255) / 256, 256, 0, stream>>>(wsum);
  prep_cw<<<K_, 256, 0, stream>>>(cw, cwh, c2);
  encode_main<<<B_ * C, 256, 0, stream>>>(x, scale, cwh, c2, partial, wsum, C, T);
  const int nblk_out = (out_size + 255) / 256;
  reduce_kernel<<<nblk_out, 256, 0, stream>>>(E, partial, wsum, cw, C, out_size);
}
